// MultiHeadLatentAttention_44195213475978
// MI455X (gfx1250) — matrix operand order repaired and checked by reading, not compiled, not run on hardware
//
#include <hip/hip_runtime.h>

typedef unsigned short u16;
typedef unsigned int   u32;
typedef __attribute__((ext_vector_type(16))) __bf16 v16bf;
typedef __attribute__((ext_vector_type(8)))  float  v8f;
typedef __attribute__((ext_vector_type(4)))  unsigned int v4u;
typedef __attribute__((ext_vector_type(8)))  int v8i;
typedef __attribute__((ext_vector_type(4)))  int v4i;

#define SEQ    2048
#define NHEADS 16
#define HDIM   128
#define MROWS  4096   // B*S

#if defined(__has_builtin)
#if __has_builtin(__builtin_amdgcn_tensor_load_to_lds) && __has_builtin(__builtin_amdgcn_s_wait_tensorcnt)
#define HAVE_TDM 1
#endif
#endif

union Frag {
  v16bf v;
  u32   d[8];
  u16   h[16];
};
union U2 { uint2 v; u16 h[4]; };

__device__ __forceinline__ u16 f2bf(float f) {
  u32 u = __float_as_uint(f);
  u32 r = u + 0x7FFFu + ((u >> 16) & 1u);   // round-to-nearest-even
  return (u16)(r >> 16);
}
__device__ __forceinline__ u32 pk2bf(float lo, float hi) {
  return (u32)f2bf(lo) | ((u32)f2bf(hi) << 16);
}

#ifdef HAVE_TDM
// 2-D TENSOR_LOAD_TO_LDS: D# per ISA 8.3/8.4.  data_size=2B, optional LDS
// row padding (pad_amount dwords appended per pad_interval dwords stored).
__device__ __forceinline__ void tdm_load_2d(
    const void* gptr, u32 lds_off,
    u32 tile_d0, u32 tile_d1, u32 stride0,
    u32 tensor_d0, u32 tensor_d1,
    u32 pad_iv_code, u32 pad_amt_code)
{
  const unsigned long long ga = (unsigned long long)(size_t)gptr;
  v4u g0; v8i g1;
  const v4i z4 = {0, 0, 0, 0};
  const v8i z8 = {0, 0, 0, 0, 0, 0, 0, 0};
  g0.x = 1u;                                              // count=1, user D#
  g0.y = lds_off;                                         // lds_addr
  g0.z = (u32)ga;                                         // global_addr[31:0]
  g0.w = ((u32)(ga >> 32) & 0x01FFFFFFu) | 0x80000000u;   // addr[56:32] | type=2
  g1[0] = (int)((1u << 16) | (1u << 20) | (pad_iv_code << 22) | (pad_amt_code << 25));
  g1[1] = (int)(tensor_d0 << 16);
  g1[2] = (int)((tensor_d0 >> 16) | (tensor_d1 << 16));
  g1[3] = (int)((tensor_d1 >> 16) | (tile_d0 << 16));
  g1[4] = (int)tile_d1;
  g1[5] = (int)stride0;
  g1[6] = 0; g1[7] = 0;
  __builtin_amdgcn_tensor_load_to_lds(g0, g1, z4, z4, z8, 0);
}
#endif

// ---------------------------------------------------------------------------
// bf16-WMMA GEMM: C(MxN) = A(MxK) * B(KxN).  128x128x32 blocks, 8 wave32.
// A tile LDS: row-major [m][k], stride 40; loaded by TDM when A is bf16.
// B tile LDS: TRANSPOSED [n][k], stride 40 -> B fragment = 2x ds_load_b128.
// B fragments software-pipelined 4 deep to hide LDS latency between WMMAs.
// EP: 0 f32 rowmajor | 1 bf16 rowmajor | 2 bf16 head-major | 3 = 2 + RoPE
// ---------------------------------------------------------------------------
template<int A_BF16, int EP>
__global__ __launch_bounds__(256) void gemm_wmma(
    const float* __restrict__ Af32, const u16* __restrict__ Abf, int lda,
    const float* __restrict__ Bw, int ldb,
    void* __restrict__ Cout, int Ndim, int Kdim)
{
  __shared__ __align__(16) u16 As[128 * 40];
  __shared__ __align__(16) u16 BsT[128 * 40];

  const int tid  = threadIdx.x;
  const int wave = tid >> 5, lane = tid & 31;
  const int mq   = lane & 15, hf = lane >> 4;
  const int bm   = blockIdx.y * 128, bn = blockIdx.x * 128;

  const v8f zacc = {0.f,0.f,0.f,0.f,0.f,0.f,0.f,0.f};
  v8f acc[8];
  #pragma unroll
  for (int i = 0; i < 8; ++i) acc[i] = zacc;

  for (int k0 = 0; k0 < Kdim; k0 += 32) {
    // ---- stage A tile (128x32) ----
    if constexpr (A_BF16) {
#ifdef HAVE_TDM
      if (wave == 0)
        tdm_load_2d(Abf + (size_t)bm * lda + k0, (u32)(size_t)&As[0],
                    /*tile*/32, 128, /*stride*/(u32)lda,
                    /*tensor*/32, 128, /*pad 4dw per 16dw*/3, 3);
#else
      {
        const int r0 = tid >> 3, c = (tid & 7) * 4;
        uint2 t4[4];
        #pragma unroll
        for (int rr = 0; rr < 4; ++rr)
          t4[rr] = *(const uint2*)(Abf + (size_t)(bm + r0 + rr * 32) * lda + k0 + c);
        #pragma unroll
        for (int rr = 0; rr < 4; ++rr)
          *(uint2*)&As[(r0 + rr * 32) * 40 + c] = t4[rr];
      }
#endif
    } else {
      const int r0 = tid >> 3, c = (tid & 7) * 4;
      float4 t4[4];
      #pragma unroll
      for (int rr = 0; rr < 4; ++rr)
        t4[rr] = *(const float4*)(Af32 + (size_t)(bm + r0 + rr * 32) * lda + k0 + c);
      #pragma unroll
      for (int rr = 0; rr < 4; ++rr) {
        uint2 p; p.x = pk2bf(t4[rr].x, t4[rr].y); p.y = pk2bf(t4[rr].z, t4[rr].w);
        *(uint2*)&As[(r0 + rr * 32) * 40 + c] = p;
      }
    }
    // ---- stage B tile (32x128) transposed: load K-row pairs, pack dwords ----
    {
      const int n0 = (tid & 31) * 4, kp = tid >> 5;   // kp in 0..7, pairs kp & kp+8
      const float* bp = Bw + (size_t)k0 * ldb + bn + n0;
      float4 a0 = *(const float4*)(bp + (size_t)(2 * kp)      * ldb);
      float4 b0 = *(const float4*)(bp + (size_t)(2 * kp + 1)  * ldb);
      float4 a1 = *(const float4*)(bp + (size_t)(2 * kp + 16) * ldb);
      float4 b1 = *(const float4*)(bp + (size_t)(2 * kp + 17) * ldb);
      const float a0v[4] = {a0.x, a0.y, a0.z, a0.w}, b0v[4] = {b0.x, b0.y, b0.z, b0.w};
      const float a1v[4] = {a1.x, a1.y, a1.z, a1.w}, b1v[4] = {b1.x, b1.y, b1.z, b1.w};
      #pragma unroll
      for (int i = 0; i < 4; ++i) {
        *(u32*)&BsT[(n0 + i) * 40 + 2 * kp]       = pk2bf(a0v[i], b0v[i]);
        *(u32*)&BsT[(n0 + i) * 40 + 2 * (kp + 8)] = pk2bf(a1v[i], b1v[i]);
      }
    }
    if (k0 + 32 < Kdim)   // global_prefetch_b8 hint for next B tile
      __builtin_prefetch(Bw + (size_t)(k0 + 32 + (tid >> 5)) * ldb + bn + (tid & 31) * 4, 0, 1);
#ifdef HAVE_TDM
    if constexpr (A_BF16) {
      if (wave == 0) __builtin_amdgcn_s_wait_tensorcnt(0);
    }
#endif
    __syncthreads();

    // ---- A fragment: 2x b128 (K contiguous 16 per wave-half) ----
    Frag af;
    ((uint4*)af.d)[0] = *(const uint4*)&As[(wave * 16 + mq) * 40 + 16 * hf];
    ((uint4*)af.d)[1] = *(const uint4*)&As[(wave * 16 + mq) * 40 + 16 * hf + 8];

    auto loadBf = [&](Frag& f, int nt) {
      const u16* p = &BsT[(nt * 16 + mq) * 40 + 16 * hf];
      ((uint4*)f.d)[0] = *(const uint4*)(p);
      ((uint4*)f.d)[1] = *(const uint4*)(p + 8);
    };
    // ---- 8 N-tiles, B fragments pipelined 4 deep ----
    Frag bfr[4];
    #pragma unroll
    for (int j = 0; j < 4; ++j) loadBf(bfr[j], j);
    #pragma unroll
    for (int nt = 0; nt < 8; ++nt) {
      acc[nt] = __builtin_amdgcn_wmma_f32_16x16x32_bf16(
          false, af.v, false, bfr[nt & 3].v, (short)0, acc[nt], false, false);
      if (nt < 4) loadBf(bfr[nt & 3], nt + 4);
    }
    __syncthreads();
  }

  // ---- epilogue ----
  #pragma unroll
  for (int nt = 0; nt < 8; ++nt) {
    #pragma unroll
    for (int i = 0; i < 8; ++i) {
      const int row = bm + wave * 16 + i + 8 * hf;   // C/D: M = i + 8*half
      const int col = bn + nt * 16 + mq;             // N = lane&15
      float v = acc[nt][i];
      if constexpr (EP == 0) {
        ((float*)Cout)[(size_t)row * Ndim + col] = v;
      } else if constexpr (EP == 1) {
        ((u16*)Cout)[(size_t)row * Ndim + col] = f2bf(v);
      } else {
        const int s = row & (SEQ - 1), bb = row >> 11;
        const int hh = col >> 7, d = col & 127;
        if constexpr (EP == 3) {
          // RoPE: cos/sin idx = d mod 64 (concat(freqs,freqs)); interleaved rotate_half
          const float vp  = __shfl_xor(v, 1, 32);
          const float ang = (float)s * exp2f(-(float)(d & 63) * 0.20762050593046468f);
          const float sn = __sinf(ang), cs = __cosf(ang);
          const float rot = (d & 1) ? vp : -vp;
          v = v * cs + rot * sn;
        }
        ((u16*)Cout)[(((size_t)(bb * NHEADS + hh)) * SEQ + s) * HDIM + d] = f2bf(v);
      }
    }
  }
}

// ---------------------------------------------------------------------------
// Flash attention: WG per (b,h,128 q-rows), 8 waves x 16 q-rows.
// K chunk (32x128 bf16) loaded row-major via TDM (LDS pad -> 136 stride);
// V chunk staged transposed [d][t]; all WMMA fragments software-pipelined.
// ---------------------------------------------------------------------------
__global__ __launch_bounds__(256) void mla_attn(
    const u16* __restrict__ Q, const u16* __restrict__ Kg,
    const u16* __restrict__ Vg, u16* __restrict__ Aout)
{
  __shared__ __align__(16) u16 Ksh[32 * 136];   // [t][d] + 8 pad (TDM pad feature)
  __shared__ __align__(16) u16 VT[128 * 40];    // [d][t] transposed
  __shared__ __align__(16) u16 Psh[8][16 * 40]; // per-wave P re-layout staging

  const int tid  = threadIdx.x;
  const int wave = tid >> 5, lane = tid & 31;
  const int mq   = lane & 15, hf = lane >> 4;
  const int bh   = blockIdx.y;
  const int qbase = blockIdx.x * 128;
  const int q0    = qbase + wave * 16;
  const size_t base = (size_t)bh * SEQ * HDIM;
  const float scale = 0.08838834764831845f;  // 1/sqrt(128)

  // Q fragments: 4 K-slices of 32 over HEAD_DIM (2x b128 global loads each)
  Frag qf[4];
  #pragma unroll
  for (int sl = 0; sl < 4; ++sl) {
    const u16* qp = Q + base + (size_t)(q0 + mq) * HDIM + sl * 32 + 16 * hf;
    ((uint4*)qf[sl].d)[0] = *(const uint4*)(qp);
    ((uint4*)qf[sl].d)[1] = *(const uint4*)(qp + 8);
  }

  const v8f zacc = {0.f,0.f,0.f,0.f,0.f,0.f,0.f,0.f};
  v8f Ot[8];
  #pragma unroll
  for (int i = 0; i < 8; ++i) Ot[i] = zacc;
  float ms[8], ls[8];
  #pragma unroll
  for (int i = 0; i < 8; ++i) { ms[i] = -1e30f; ls[i] = 0.f; }

  const int nchunk = qbase / 32 + 4;   // causal: only t <= qbase+127
  for (int ch = 0; ch < nchunk; ++ch) {
    const int t0 = ch * 32;
    // ---- V chunk: transposed staging with t-row-pair dword packing ----
    {
      const int d0 = (tid & 31) * 4, tp = tid >> 5;
      const u16* vp = Vg + base + (size_t)t0 * HDIM + d0;
      U2 a0, b0, a1, b1;
      a0.v = *(const uint2*)(vp + (size_t)(2 * tp)      * HDIM);
      b0.v = *(const uint2*)(vp + (size_t)(2 * tp + 1)  * HDIM);
      a1.v = *(const uint2*)(vp + (size_t)(2 * tp + 16) * HDIM);
      b1.v = *(const uint2*)(vp + (size_t)(2 * tp + 17) * HDIM);
      #pragma unroll
      for (int i = 0; i < 4; ++i) {
        *(u32*)&VT[(d0 + i) * 40 + 2 * tp]       = (u32)a0.h[i] | ((u32)b0.h[i] << 16);
        *(u32*)&VT[(d0 + i) * 40 + 2 * (tp + 8)] = (u32)a1.h[i] | ((u32)b1.h[i] << 16);
      }
    }
    // ---- K chunk ----
#ifdef HAVE_TDM
    if (wave == 0) {
      tdm_load_2d(Kg + base + (size_t)t0 * HDIM, (u32)(size_t)&Ksh[0],
                  /*tile*/128, 32, /*stride*/128,
                  /*tensor*/128, 2048, /*pad 4dw per 64dw*/5, 3);
      __builtin_amdgcn_s_wait_tensorcnt(0);
    }
#else
    {
      const int r = tid >> 3, c = (tid & 7) * 16;
      const u16* ks = Kg + base + (size_t)(t0 + r) * HDIM + c;
      uint4 ka = *(const uint4*)ks;
      uint4 kb = *(const uint4*)(ks + 8);
      *(uint4*)&Ksh[r * 136 + c]     = ka;
      *(uint4*)&Ksh[r * 136 + c + 8] = kb;
    }
#endif
    __syncthreads();

    // ---- scores S = Q*K^T : 16x32 (2 C tiles), slices pipelined 2 deep ----
    auto loadKf = [&](Frag& f, int row, int sl) {
      const u16* p = &Ksh[row * 136 + sl * 32 + 16 * hf];
      ((uint4*)f.d)[0] = *(const uint4*)(p);
      ((uint4*)f.d)[1] = *(const uint4*)(p + 8);
    };
    v8f S0 = zacc, S1 = zacc;
    Frag ka[2], kb[2];
    loadKf(ka[0], mq, 0);
    loadKf(kb[0], 16 + mq, 0);
    #pragma unroll
    for (int sl = 0; sl < 4; ++sl) {
      const int cur = sl & 1, nxt = cur ^ 1;
      if (sl < 3) { loadKf(ka[nxt], mq, sl + 1); loadKf(kb[nxt], 16 + mq, sl + 1); }
      S0 = __builtin_amdgcn_wmma_f32_16x16x32_bf16(false, qf[sl].v, false, ka[cur].v, (short)0, S0, false, false);
      S1 = __builtin_amdgcn_wmma_f32_16x16x32_bf16(false, qf[sl].v, false, kb[cur].v, (short)0, S1, false, false);
    }

    // ---- online softmax over the 32-wide chunk ----
    #pragma unroll
    for (int i = 0; i < 8; ++i) {
      const int rg = q0 + i + 8 * hf;
      float s0 = S0[i] * scale; if (t0 + mq > rg)      s0 = -1e9f;
      float s1 = S1[i] * scale; if (t0 + 16 + mq > rg) s1 = -1e9f;
      float v = fmaxf(s0, s1);
      #pragma unroll
      for (int off = 1; off < 16; off <<= 1) v = fmaxf(v, __shfl_xor(v, off, 32));
      const float mnew  = fmaxf(ms[i], v);
      const float alpha = __expf(ms[i] - mnew);
      const float p0 = __expf(s0 - mnew), p1 = __expf(s1 - mnew);
      float rs = p0 + p1;
      #pragma unroll
      for (int off = 1; off < 16; off <<= 1) rs += __shfl_xor(rs, off, 32);
      ls[i] = ls[i] * alpha + rs;
      ms[i] = mnew;
      #pragma unroll
      for (int dt = 0; dt < 8; ++dt) Ot[dt][i] *= alpha;
      Psh[wave][(i + 8 * hf) * 40 + mq]      = f2bf(p0);
      Psh[wave][(i + 8 * hf) * 40 + 16 + mq] = f2bf(p1);
    }

    asm volatile("s_wait_dscnt 0" ::: "memory");   // per-wave LDS RAW (P stage)

    Frag pf;   // P fragment (A layout): 2x b128
    ((uint4*)pf.d)[0] = *(const uint4*)&Psh[wave][mq * 40 + 16 * hf];
    ((uint4*)pf.d)[1] = *(const uint4*)&Psh[wave][mq * 40 + 16 * hf + 8];

    // ---- O += P * V (8 d-tiles), V fragments pipelined 4 deep ----
    auto loadVf = [&](Frag& f, int dt) {
      const u16* p = &VT[(dt * 16 + mq) * 40 + 16 * hf];
      ((uint4*)f.d)[0] = *(const uint4*)(p);
      ((uint4*)f.d)[1] = *(const uint4*)(p + 8);
    };
    Frag vfr[4];
    #pragma unroll
    for (int j = 0; j < 4; ++j) loadVf(vfr[j], j);
    #pragma unroll
    for (int dt = 0; dt < 8; ++dt) {
      Ot[dt] = __builtin_amdgcn_wmma_f32_16x16x32_bf16(
          false, pf.v, false, vfr[dt & 3].v, (short)0, Ot[dt], false, false);
      if (dt < 4) loadVf(vfr[dt & 3], dt + 4);
    }
    __syncthreads();
  }

  // ---- write attention out row-major (b*S+s, h*128+d) bf16 for Wo GEMM ----
  const int bb = bh >> 4, hh = bh & 15;
  #pragma unroll
  for (int dt = 0; dt < 8; ++dt) {
    #pragma unroll
    for (int i = 0; i < 8; ++i) {
      const int s = q0 + i + 8 * hf;
      const size_t row = (size_t)bb * SEQ + s;
      const int col = hh * HDIM + dt * 16 + mq;
      Aout[row * 2048 + col] = f2bf(Ot[dt][i] / ls[i]);
    }
  }
}

// ---------------------------------------------------------------------------
extern "C" void kernel_launch(void* const* d_in, const int* in_sizes, int n_in,
                              void* d_out, int out_size, void* d_ws, size_t ws_size,
                              hipStream_t stream)
{
  const float* x    = (const float*)d_in[0];
  // d_in[1] = attn_mask (causal tril) -- implemented directly in the kernel
  const float* Wq   = (const float*)d_in[2];
  const float* Wdkv = (const float*)d_in[3];
  const float* Wuk  = (const float*)d_in[4];
  const float* Wuv  = (const float*)d_in[5];
  const float* Wo   = (const float*)d_in[6];
  float* out = (float*)d_out;

  char* w = (char*)d_ws;
  const size_t QB = (size_t)MROWS * 2048 * sizeof(u16);
  u16* qb    = (u16*)w; w += QB;
  u16* kb    = (u16*)w; w += QB;
  u16* vb    = (u16*)w; w += QB;
  u16* attnb = (u16*)w; w += QB;
  u16* dkvb  = (u16*)w; w += (size_t)MROWS * 1024 * sizeof(u16);

  const dim3 blk(256);
  // Q = rope(x @ Wq)  -> bf16 head-major
  gemm_wmma<0,3><<<dim3(16,32), blk, 0, stream>>>(x, nullptr, 2048, Wq,   2048, qb,   2048, 2048);
  // dkv = x @ Wdkv    -> bf16 row-major
  gemm_wmma<0,1><<<dim3( 8,32), blk, 0, stream>>>(x, nullptr, 2048, Wdkv, 1024, dkvb, 1024, 2048);
  // K = rope(d_k @ Wuk) -> bf16 head-major
  gemm_wmma<1,3><<<dim3(16,32), blk, 0, stream>>>(nullptr, dkvb,       1024, Wuk, 2048, kb, 2048, 512);
  // V = d_v @ Wuv       -> bf16 head-major
  gemm_wmma<1,2><<<dim3(16,32), blk, 0, stream>>>(nullptr, dkvb + 512, 1024, Wuv, 2048, vb, 2048, 512);
  // causal attention
  mla_attn<<<dim3(16,32), blk, 0, stream>>>(qb, kb, vb, attnb);
  // out = attn @ Wo -> f32
  gemm_wmma<1,0><<<dim3(16,32), blk, 0, stream>>>(nullptr, attnb, 2048, Wo, 2048, out, 2048, 2048);
}
